// ResidualVectorQuantizerGCN_72670846648464
// MI455X (gfx1250) — matrix operand order repaired and checked by reading, not compiled, not run on hardware
//
#include <hip/hip_runtime.h>

typedef _Float16 v8h  __attribute__((ext_vector_type(8)));
typedef _Float16 v16h __attribute__((ext_vector_type(16)));
typedef float    v8f  __attribute__((ext_vector_type(8)));
typedef int      v4i  __attribute__((vector_size(16)));
typedef __attribute__((address_space(1))) v4i* as1_v4i_ptr;
typedef __attribute__((address_space(3))) v4i* as3_v4i_ptr;

#define V_NODES 25
#define DIM 128
#define NQ 4
#define NE 1024
#define ALPHA 0.1f
#define SLOPE 0.2f
#define LN_EPS 1e-5f
#define CHUNK 64
#define NCHUNK (NE / CHUNK)
#define GPB 4                 // graphs per block (1 per wave32)
#define ZQ_ELEMS (16 * 200 * 25 * 128)
#define LOSS_OFF ((size_t)ZQ_ELEMS)
#define IDX_OFF  ((size_t)ZQ_ELEMS + 1)
// loss scale: (1+beta)=1.25 over 4 stages * 80000 vectors * 128 dims
#define LOSS_SCALE (1.25f / 40960000.0f)

#if __has_builtin(__builtin_amdgcn_global_load_async_to_lds_b128)
#define ASYNC_CP 1
#endif

__device__ __forceinline__ void cp16_async(void* lds, const void* g) {
#ifdef ASYNC_CP
  __builtin_amdgcn_global_load_async_to_lds_b128(
      (as1_v4i_ptr)(void*)g, (as3_v4i_ptr)(void*)lds, 0, 0);
#else
  *(v8h*)lds = *(const v8h*)g;
#endif
}

__device__ __forceinline__ void async_wait0() {
#ifdef ASYNC_CP
#if __has_builtin(__builtin_amdgcn_s_wait_asynccnt)
  __builtin_amdgcn_s_wait_asynccnt(0);
#else
  asm volatile("s_wait_asynccnt 0" ::: "memory");
#endif
#endif
}

// A-fragment (16x32 f16, M x K): lane holds row m = l16, K k0..k0+15 contiguous
// (+16 for lanes 16-31), matching the contiguous B fragment.
__device__ __forceinline__ v16h frag_a(const _Float16* row, int k0, int hf8) {
  const _Float16* p = row + k0 + 2 * hf8;
  v8h lo = *(const v8h*)p;
  v8h hi = *(const v8h*)(p + 8);
  return __builtin_shufflevector(lo, hi, 0, 1, 2, 3, 4, 5, 6, 7, 8, 9, 10, 11, 12, 13, 14, 15);
}

__device__ __forceinline__ v8f wmma_f16(v16h a, v16h b, v8f c) {
  return __builtin_amdgcn_wmma_f32_16x16x32_f16(false, a, false, b, (short)0, c, false, false);
}

// ---------------- prep: f32->f16 codebooks + gcn weights, -0.5*||w||^2, zero loss ----------------
__global__ __launch_bounds__(128) void rvq_prep(const float* __restrict__ emb,
                                                const float* __restrict__ gcn_w,
                                                _Float16* __restrict__ embh,
                                                _Float16* __restrict__ gcnwh,
                                                float* __restrict__ nhw,
                                                float* __restrict__ out) {
  int b = blockIdx.x;
  int t = threadIdx.x;
  if (b < NQ * NE) {
    float v = emb[(size_t)b * DIM + t];
    embh[(size_t)b * DIM + t] = (_Float16)v;
    float s = v * v;
    __shared__ float red[4];
    for (int o = 16; o; o >>= 1) s += __shfl_xor(s, o, 32);
    if ((t & 31) == 0) red[t >> 5] = s;
    __syncthreads();
    if (t == 0) nhw[b] = -0.5f * (red[0] + red[1] + red[2] + red[3]);
  } else if (b < NQ * NE + DIM) {
    int r = b - NQ * NE;
    gcnwh[r * DIM + t] = (_Float16)gcn_w[r * DIM + t];
  } else {
    if (t == 0) out[LOSS_OFF] = 0.0f;
  }
}

// ---------------- main fused RVQ kernel ----------------
__global__ __launch_bounds__(128, 1) void rvq_main(
    const float* __restrict__ z, const float* __restrict__ emb,
    const float* __restrict__ A_norm, const float* __restrict__ gcn_b,
    const float* __restrict__ ln_s, const float* __restrict__ ln_b,
    const _Float16* __restrict__ embh, const _Float16* __restrict__ gcnwh,
    const float* __restrict__ nhw, float* __restrict__ out) {
  __shared__ __align__(32) float    s_res[GPB][V_NODES * DIM];   // residual f32
  __shared__ __align__(32) _Float16 s_xh[GPB][32 * DIM];         // neigh/refined f16 (padded rows)
  __shared__ __align__(32) union MixU {
    float    scr[GPB][V_NODES * DIM];    // linear output (steps 2-3)
    _Float16 cb[2][CHUNK * DIM];         // codebook double buffer (step 4)
  } u;
  __shared__ __align__(32) _Float16 s_gw[DIM * DIM];             // gcn_w f16
  __shared__ float s_A[V_NODES * V_NODES];
  __shared__ int   s_idx[GPB][32];
  __shared__ float s_lred[GPB];

  const int tid  = threadIdx.x;
  const int wave = tid >> 5;
  const int lane = tid & 31;
  const int hf   = lane >> 4;   // 0: lanes 0-15, 1: lanes 16-31
  const int l16  = lane & 15;
  const int g    = blockIdx.x * GPB + wave;   // graph id, 0..3199

  // load gcn weights + adjacency, init residual = z, zero xh pad rows
  for (int i = tid; i < DIM * DIM / 8; i += 128)
    *(v8h*)(s_gw + i * 8) = *(const v8h*)(gcnwh + i * 8);
  for (int i = tid; i < V_NODES * V_NODES; i += 128) s_A[i] = A_norm[i];
  {
    const float* zg = z + (size_t)g * V_NODES * DIM;
    float* res = s_res[wave];
    for (int i = lane; i < V_NODES * DIM; i += 32) res[i] = zg[i];
    _Float16* xh = s_xh[wave];
    for (int i = lane; i < 32 * DIM; i += 32) xh[i] = (_Float16)0.0f;
  }
  __syncthreads();

  float lossacc = 0.0f;

  for (int k = 0; k < NQ; ++k) {
    float*    res = s_res[wave];
    _Float16* xh  = s_xh[wave];
    float*    scr = u.scr[wave];

    // ---- step 1: ring-GCN aggregation: xh = f16(A_norm @ residual) ----
    for (int r = 0; r < V_NODES; ++r) {
      float a0 = 0.f, a1 = 0.f, a2 = 0.f, a3 = 0.f;
      const float* Ar = s_A + r * V_NODES;
      for (int w = 0; w < V_NODES; ++w) {
        float aw = Ar[w];
        const float* rw = res + w * DIM + lane;
        a0 += aw * rw[0]; a1 += aw * rw[32]; a2 += aw * rw[64]; a3 += aw * rw[96];
      }
      _Float16* xr = xh + r * DIM + lane;
      xr[0] = (_Float16)a0; xr[32] = (_Float16)a1; xr[64] = (_Float16)a2; xr[96] = (_Float16)a3;
    }
    __syncthreads();

    // ---- step 2: linear via WMMA: scr[row][j] = gcn_w[j,:].neigh[row,:] + b[j] ----
    for (int nt = 0; nt < 2; ++nt) {
      const _Float16* brow = xh + (nt * 16 + l16) * DIM;
      for (int mt = 0; mt < 8; ++mt) {
        v8f c = *(const v8f*)(gcn_b + mt * 16 + hf * 8);   // bias seed
        const _Float16* arow = s_gw + (mt * 16 + l16) * DIM;
#pragma unroll
        for (int kk = 0; kk < 4; ++kk)
          c = wmma_f16(frag_a(arow, kk * 32, hf * 8),
                       *(const v16h*)(brow + kk * 32 + hf * 16), c);
        int n = nt * 16 + l16;
        if (n < V_NODES) *(v8f*)(scr + n * DIM + mt * 16 + hf * 8) = c;
      }
    }
    __syncthreads();

    // ---- step 3: LeakyReLU + LayerNorm + residual blend -> xh = f16(refined) ----
    for (int r = 0; r < V_NODES; ++r) {
      float* sr = scr + r * DIM;
      float v0 = sr[lane], v1 = sr[lane + 32], v2 = sr[lane + 64], v3 = sr[lane + 96];
      v0 = v0 >= 0.f ? v0 : SLOPE * v0;  v1 = v1 >= 0.f ? v1 : SLOPE * v1;
      v2 = v2 >= 0.f ? v2 : SLOPE * v2;  v3 = v3 >= 0.f ? v3 : SLOPE * v3;
      float s = v0 + v1 + v2 + v3;
      float s2 = v0 * v0 + v1 * v1 + v2 * v2 + v3 * v3;
      for (int o = 16; o; o >>= 1) { s += __shfl_xor(s, o, 32); s2 += __shfl_xor(s2, o, 32); }
      float mu = s * (1.0f / DIM);
      float var = s2 * (1.0f / DIM) - mu * mu;
      float rs = __frsqrt_rn(var + LN_EPS);
      float* rr = res + r * DIM;
      _Float16* xr = xh + r * DIM;
      xr[lane]      = (_Float16)(rr[lane]      + ALPHA * ((v0 - mu) * rs * ln_s[lane]      + ln_b[lane]));
      xr[lane + 32] = (_Float16)(rr[lane + 32] + ALPHA * ((v1 - mu) * rs * ln_s[lane + 32] + ln_b[lane + 32]));
      xr[lane + 64] = (_Float16)(rr[lane + 64] + ALPHA * ((v2 - mu) * rs * ln_s[lane + 64] + ln_b[lane + 64]));
      xr[lane + 96] = (_Float16)(rr[lane + 96] + ALPHA * ((v3 - mu) * rs * ln_s[lane + 96] + ln_b[lane + 96]));
    }
    __syncthreads();

    // ---- step 4: codebook distance via WMMA, running argmax of (x.w - 0.5||w||^2) ----
    const _Float16* ebase = embh + (size_t)k * NE * DIM;
    const float* wnk = nhw + k * NE;
    {
      const _Float16* src = ebase;
      _Float16* dst = u.cb[0];
#pragma unroll
      for (int i = 0; i < 8; ++i) cp16_async(dst + tid * 64 + i * 8, src + tid * 64 + i * 8);
      async_wait0();
    }
    __syncthreads();

    float bv0 = -1e30f, bv1 = -1e30f;
    int bi0 = 0, bi1 = 0;
    for (int c = 0; c < NCHUNK; ++c) {
      if (c + 1 < NCHUNK) {   // async prefetch next chunk into other buffer
        const _Float16* src = ebase + (size_t)(c + 1) * CHUNK * DIM;
        _Float16* dst = u.cb[(c + 1) & 1];
#pragma unroll
        for (int i = 0; i < 8; ++i) cp16_async(dst + tid * 64 + i * 8, src + tid * 64 + i * 8);
      }
      const _Float16* cb = u.cb[c & 1];
      for (int mt = 0; mt < 4; ++mt) {
        const _Float16* arow = cb + (mt * 16 + l16) * DIM;
        int embase = c * CHUNK + mt * 16 + hf * 8;
        v8f cseed = *(const v8f*)(wnk + embase);    // -0.5*||w||^2 seed
        // node rows 0-15
        {
          const _Float16* brow = xh + l16 * DIM;
          v8f acc = cseed;
#pragma unroll
          for (int kk = 0; kk < 4; ++kk)
            acc = wmma_f16(frag_a(arow, kk * 32, hf * 8),
                           *(const v16h*)(brow + kk * 32 + hf * 16), acc);
#pragma unroll
          for (int rj = 0; rj < 8; ++rj) {
            float vv = acc[rj];
            if (vv > bv0) { bv0 = vv; bi0 = embase + rj; }
          }
        }
        // node rows 16-24 (+pad)
        {
          const _Float16* brow = xh + (16 + l16) * DIM;
          v8f acc = cseed;
#pragma unroll
          for (int kk = 0; kk < 4; ++kk)
            acc = wmma_f16(frag_a(arow, kk * 32, hf * 8),
                           *(const v16h*)(brow + kk * 32 + hf * 16), acc);
#pragma unroll
          for (int rj = 0; rj < 8; ++rj) {
            float vv = acc[rj];
            if (vv > bv1) { bv1 = vv; bi1 = embase + rj; }
          }
        }
      }
      async_wait0();
      __syncthreads();
    }

    // combine lane L with lane L+16 (disjoint entry halves), prefer smaller idx on ties
    {
      float o0 = __shfl_xor(bv0, 16, 32); int oi0 = __shfl_xor(bi0, 16, 32);
      if (o0 > bv0 || (o0 == bv0 && oi0 < bi0)) { bv0 = o0; bi0 = oi0; }
      float o1 = __shfl_xor(bv1, 16, 32); int oi1 = __shfl_xor(bi1, 16, 32);
      if (o1 > bv1 || (o1 == bv1 && oi1 < bi1)) { bv1 = o1; bi1 = oi1; }
      if (hf == 0) {
        s_idx[wave][l16] = bi0;
        if (l16 < V_NODES - 16) s_idx[wave][16 + l16] = bi1;
        float* oidx = out + IDX_OFF;
        oidx[((size_t)g * V_NODES + l16) * NQ + k] = (float)bi0;
        if (l16 < V_NODES - 16)
          oidx[((size_t)g * V_NODES + 16 + l16) * NQ + k] = (float)bi1;
      }
    }
    __syncthreads();

    // ---- step 5: residual -= q ; loss += ||q - residual_old||^2 ----
    const float* embk = emb + (size_t)k * NE * DIM;
    for (int r = 0; r < V_NODES; ++r) {
      int e = s_idx[wave][r];
      const float* wrow = embk + (size_t)e * DIM;
      float* rr = res + r * DIM;
#pragma unroll
      for (int j = 0; j < 4; ++j) {
        int d = lane + 32 * j;
        float w = wrow[d];
        float diff = w - rr[d];
        lossacc += diff * diff;
        rr[d] -= w;
      }
    }
    __syncthreads();
  }

  // zq = z - residual_final  (== straight-through cum)
  {
    const float* zg = z + (size_t)g * V_NODES * DIM;
    float* zq = out + (size_t)g * V_NODES * DIM;
    float* res = s_res[wave];
    for (int i = lane; i < V_NODES * DIM; i += 32) zq[i] = zg[i] - res[i];
  }
  // loss reduction: lanes -> waves -> one atomic per block
  for (int o = 16; o; o >>= 1) lossacc += __shfl_xor(lossacc, o, 32);
  if (lane == 0) s_lred[wave] = lossacc;
  __syncthreads();
  if (tid == 0) {
    float t = (s_lred[0] + s_lred[1] + s_lred[2] + s_lred[3]) * LOSS_SCALE;
    atomicAdd(out + LOSS_OFF, t);
  }
}

extern "C" void kernel_launch(void* const* d_in, const int* in_sizes, int n_in,
                              void* d_out, int out_size, void* d_ws, size_t ws_size,
                              hipStream_t stream) {
  (void)in_sizes; (void)n_in; (void)out_size; (void)ws_size;
  const float* z      = (const float*)d_in[0];
  const float* emb    = (const float*)d_in[1];
  const float* A_norm = (const float*)d_in[2];
  const float* gcn_w  = (const float*)d_in[3];
  const float* gcn_b  = (const float*)d_in[4];
  const float* ln_s   = (const float*)d_in[5];
  const float* ln_b   = (const float*)d_in[6];
  float* out = (float*)d_out;

  _Float16* embh  = (_Float16*)d_ws;                                   // 1 MB
  _Float16* gcnwh = (_Float16*)((char*)d_ws + (1 << 20));              // 32 KB
  float*    nhw   = (float*)((char*)d_ws + (1 << 20) + (1 << 15));     // 16 KB

  rvq_prep<<<NQ * NE + DIM + 1, 128, 0, stream>>>(emb, gcn_w, embh, gcnwh, nhw, out);
  rvq_main<<<3200 / GPB, 128, 0, stream>>>(z, emb, A_norm, gcn_b, ln_s, ln_b,
                                           embh, gcnwh, nhw, out);
}
